// Attention_73031623901249
// MI455X (gfx1250) — matrix operand order repaired and checked by reading, not compiled, not run on hardware
//
#include <hip/hip_runtime.h>
#include <hip/hip_bf16.h>

// MI455X / gfx1250, wave32. All matrix math via v_wmma_f32_16x16x32_bf16.
// Workspace (bf16): qh (bh,s,d) | kh (bh,s,d) | vht (bh,d,s) | attn (b,s,h*d)
// each B*H*S*D = 16,777,216 elems (~134 MB total -> resident in 192 MB L2).
// Softmax runs in the exp2 domain: qh carries (1/sqrt(D))*log2(e).

typedef __bf16 bf16_t;
typedef __attribute__((ext_vector_type(16))) __bf16 v16bf;
typedef __attribute__((ext_vector_type(8)))  __bf16 v8bf;
typedef __attribute__((ext_vector_type(4)))  __bf16 v4bf;
typedef __attribute__((ext_vector_type(8)))  float  v8f;

constexpr int B_  = 2;
constexpr int S_  = 2048;
constexpr int D_  = 512;     // per-head dim (full D in this model)
constexpr int H_  = 8;
constexpr int HD_ = H_ * D_; // 4096

__device__ __forceinline__ float fast_exp2(float x) {
  return __builtin_amdgcn_exp2f(x);    // v_exp_f32
}

__device__ __forceinline__ v8f vzero() {
  v8f z;
#pragma unroll
  for (int i = 0; i < 8; ++i) z[i] = 0.f;
  return z;
}

// A-operand fragment, 16x32 bf16, source row-major (M,K).  ISA 7.12.2:
// lane=(half<<4)|m ; half=0 holds K {0..7}u{16..23}, half=1 holds {8..15}u{24..31}.
template <typename T>
__device__ __forceinline__ v16bf load_frag_a(const T* base, int ldk, int lane) {
  const int row  = lane & 15;
  const int half = (lane >> 4) & 1;
  const T* p = base + (size_t)row * ldk + half * 8;
  v16bf f;
#pragma unroll
  for (int i = 0; i < 8; ++i) f[i] = (bf16_t)p[i];
#pragma unroll
  for (int i = 0; i < 8; ++i) f[8 + i] = (bf16_t)p[16 + i];
  return f;
}

// B-operand fragment, 32x16 bf16, source N-major (N,K) (B transposed: rows=N,
// cols=K).  Same K order as A: half=0 holds K {0..7}u{16..23},
// half=1 holds {8..15}u{24..31}.
template <typename T>
__device__ __forceinline__ v16bf load_frag_b(const T* base, int ldk, int lane) {
  const int col  = lane & 15;
  const int half = (lane >> 4) & 1;
  const T* p = base + (size_t)col * ldk + half * 8;
  v16bf f;
#pragma unroll
  for (int i = 0; i < 8; ++i) f[i] = (bf16_t)p[i];
#pragma unroll
  for (int i = 0; i < 8; ++i) f[8 + i] = (bf16_t)p[16 + i];
  return f;
}

__device__ __forceinline__ v8f wmma_bf16(v16bf a, v16bf b, v8f c) {
  return __builtin_amdgcn_wmma_f32_16x16x32_bf16(false, a, false, b, (short)0, c,
                                                 false, false);
}

// ---------------------------------------------------------------------------
// Kernel 1: qh/kh = RoPE(q @ W)  (scale*log2e folded into qh), vht = (q@Wv)^T.
// 64x64 tile per 128-thread WG; K-loop step 32; vectorized staging.
// ---------------------------------------------------------------------------
__global__ __launch_bounds__(128)
void qkv_rope_kernel(const float* __restrict__ q,
                     const float* __restrict__ Wq,
                     const float* __restrict__ Wk,
                     const float* __restrict__ Wv,
                     bf16_t* __restrict__ qh,
                     bf16_t* __restrict__ kh,
                     bf16_t* __restrict__ vht) {
  const int tid  = threadIdx.x;
  const int lane = tid & 31;
  const int wave = tid >> 5;                 // 0..3, rows wave*16..+15
  const int m0   = blockIdx.x * 64;          // S tile
  const int n0   = blockIdx.y * 64;          // D tile
  const int z    = blockIdx.z;               // 3 * B*H
  const int p    = z / (B_ * H_);            // 0=q,1=k,2=v
  const int bh   = z % (B_ * H_);

  const float* W = (p == 0 ? Wq : (p == 1 ? Wk : Wv)) + (size_t)(bh % H_) * D_ * D_;
  const float* A = q + (size_t)(bh / H_) * S_ * D_;

  constexpr int LT = 40;                     // padded K-stride (multiple of 8)
  __shared__ bf16_t sA[64 * LT];
  __shared__ bf16_t sB[64 * LT];

  v8f acc[4];
#pragma unroll
  for (int t = 0; t < 4; ++t) acc[t] = vzero();

  for (int k0 = 0; k0 < D_; k0 += 32) {
    // A tile 64x32 f32 -> bf16, float4 loads, v4bf stores (fixed trip count)
#pragma unroll
    for (int it = 0; it < 4; ++it) {
      const int i  = tid + it * 128;         // [0,512)
      const int r  = i >> 3;
      const int c4 = (i & 7) * 4;
      const float4 v = *(const float4*)&A[(size_t)(m0 + r) * D_ + k0 + c4];
      v4bf pk;
      pk[0] = (bf16_t)v.x; pk[1] = (bf16_t)v.y;
      pk[2] = (bf16_t)v.z; pk[3] = (bf16_t)v.w;
      *(v4bf*)&sA[r * LT + c4] = pk;
    }
    // W tile 32x64 f32, coalesced float4 reads, transposed (N-major) stores
#pragma unroll
    for (int it = 0; it < 4; ++it) {
      const int i  = tid + it * 128;         // [0,512)
      const int kk = i >> 4;                 // [0,32)
      const int n4 = (i & 15) * 4;
      const float4 v = *(const float4*)&W[(size_t)(k0 + kk) * D_ + n0 + n4];
      sB[(n4 + 0) * LT + kk] = (bf16_t)v.x;
      sB[(n4 + 1) * LT + kk] = (bf16_t)v.y;
      sB[(n4 + 2) * LT + kk] = (bf16_t)v.z;
      sB[(n4 + 3) * LT + kk] = (bf16_t)v.w;
    }
    __syncthreads();
    const v16bf af = load_frag_a(&sA[wave * 16 * LT], LT, lane);
#pragma unroll
    for (int t = 0; t < 4; ++t) {
      const v16bf bfg = load_frag_b(&sB[t * 16 * LT], LT, lane);
      acc[t] = wmma_bf16(af, bfg, acc[t]);
    }
    __syncthreads();
  }

  const int half  = lane >> 4;
  const int col16 = lane & 15;

  if (p == 2) {
    // V: store transposed, vht[bh][d][s]; each lane writes one 16B chunk.
    bf16_t* out = vht + (size_t)bh * D_ * S_;
    const int srow0 = m0 + wave * 16 + half * 8;
#pragma unroll
    for (int t = 0; t < 4; ++t) {
      const int col = n0 + t * 16 + col16;
      v8bf pk;
#pragma unroll
      for (int r = 0; r < 8; ++r) pk[r] = (bf16_t)acc[t][r];
      *(v8bf*)&out[(size_t)col * S_ + srow0] = pk;
    }
  } else {
    // Q/K: RoPE in f32, then row-major (s,d) bf16 store.
    bf16_t* out = (p == 0 ? qh : kh) + (size_t)bh * S_ * D_;
    // q gets (1/sqrt(512)) * log2(e) so the softmax runs on v_exp_f32 (exp2).
    const float scale =
        (p == 0) ? (0.04419417382415922f * 1.4426950408889634f) : 1.0f;
#pragma unroll
    for (int t = 0; t < 4; ++t) {
      const int col = n0 + t * 16 + col16;
      // 10000^(-2j/512) = exp2(-(2j) * log2(10000)/512)
      const float inv_freq =
          fast_exp2(-(float)(col & ~1) * (13.287712379549449f / (float)D_));
      const bool even = (col & 1) == 0;
#pragma unroll
      for (int r = 0; r < 8; ++r) {
        const int srow = m0 + wave * 16 + half * 8 + r;
        float sn, cs;
        __sincosf((float)srow * inv_freq, &sn, &cs);
        const float x  = acc[t][r];
        const float xo = __shfl_xor(x, 1, 32);   // partner column (col ^ 1)
        const float rr = even ? (x * cs - xo * sn) : (xo * sn + x * cs);
        out[(size_t)srow * D_ + col] = (bf16_t)(rr * scale);
      }
    }
  }
}

// ---------------------------------------------------------------------------
// Kernel 2: causal flash attention per (b,h). 64-query tile per 128-thread WG;
// each wave owns 16 rows x full 512-wide output (32 f32 accumulators).
// K and V^T operands stream straight from global (L2-resident); only the
// 16x32 P tile round-trips a per-wave LDS slab (C-layout -> A-layout).
// Scores arrive pre-scaled by log2(e): softmax uses native exp2 (v_exp_f32).
// ---------------------------------------------------------------------------
__global__ __launch_bounds__(128)
void flash_attn_kernel(const bf16_t* __restrict__ qh,
                       const bf16_t* __restrict__ kh,
                       const bf16_t* __restrict__ vht,
                       bf16_t* __restrict__ attn) {
  const int tid  = threadIdx.x;
  const int lane = tid & 31;
  const int wave = tid >> 5;            // 0..3
  const int qt   = blockIdx.x;          // S/64 tiles
  const int bh   = blockIdx.y;          // B*H
  const int b    = bh / H_;
  const int h    = bh % H_;
  const int q0   = qt * 64;

  constexpr int LP = 40;                // P key-stride (multiple of 8)
  __shared__ bf16_t sP[4 * 16 * LP];    // per-wave 16x32 P tile : 5 KB

  const bf16_t* Kbh = kh  + (size_t)bh * S_ * D_;   // (key, d)
  const bf16_t* Vbh = vht + (size_t)bh * D_ * S_;   // (d, key)

  // Preload this wave's 16 Q rows as 16 A-fragments (full K = 512).
  const bf16_t* Q = qh + ((size_t)bh * S_ + q0 + wave * 16) * D_;
  v16bf qf[16];
#pragma unroll
  for (int ks = 0; ks < 16; ++ks) qf[ks] = load_frag_a(Q + ks * 32, D_, lane);

  v8f o[32];
#pragma unroll
  for (int t = 0; t < 32; ++t) o[t] = vzero();
  float mrow[8], lrow[8];
#pragma unroll
  for (int r = 0; r < 8; ++r) { mrow[r] = -1e30f; lrow[r] = 0.f; }

  const int half  = lane >> 4;
  const int col16 = lane & 15;
  const int nkb   = qt * 2 + 2;         // causal: key blocks with k0 <= q0+63

  for (int kb = 0; kb < nkb; ++kb) {
    const int k0 = kb * 32;

    if (kb + 1 < nkb) {                 // prefetch next K / V^T block into L2/L0
      const int kn = k0 + 32;
      __builtin_prefetch(Kbh + (size_t)kn * D_ + (size_t)lane * 512, 0, 1);
      __builtin_prefetch(Vbh + (size_t)(lane * 16) * S_ + kn, 0, 1);
    }

    // Scores S[16x32] = Q . K^T ; K rows (key,d) are the N-major B layout.
    v8f s0 = vzero(), s1 = vzero();
#pragma unroll
    for (int ks = 0; ks < 16; ++ks) {
      const v16bf b0 = load_frag_b(Kbh + (size_t)k0 * D_ + ks * 32, D_, lane);
      const v16bf b1 = load_frag_b(Kbh + (size_t)(k0 + 16) * D_ + ks * 32, D_, lane);
      s0 = wmma_bf16(qf[ks], b0, s0);
      s1 = wmma_bf16(qf[ks], b1, s1);
    }

    // Causal mask + online softmax in exp2 domain
    // (each 16-lane half owns rows half*8+r).
    float alpha[8];
#pragma unroll
    for (int r = 0; r < 8; ++r) {
      const int qrow = q0 + wave * 16 + half * 8 + r;
      const float v0 = (k0 + col16 <= qrow) ? s0[r] : -1e30f;
      const float v1 = (k0 + 16 + col16 <= qrow) ? s1[r] : -1e30f;
      float mx = fmaxf(v0, v1);
#pragma unroll
      for (int msk = 1; msk < 16; msk <<= 1) mx = fmaxf(mx, __shfl_xor(mx, msk, 32));
      const float mnew = fmaxf(mrow[r], mx);
      const float a  = fast_exp2(mrow[r] - mnew);
      const float p0 = fast_exp2(v0 - mnew);
      const float p1 = fast_exp2(v1 - mnew);
      float rs = p0 + p1;
#pragma unroll
      for (int msk = 1; msk < 16; msk <<= 1) rs += __shfl_xor(rs, msk, 32);
      lrow[r] = lrow[r] * a + rs;
      mrow[r] = mnew;
      alpha[r] = a;
      s0[r] = p0;
      s1[r] = p1;
    }
#pragma unroll
    for (int t = 0; t < 32; ++t)
#pragma unroll
      for (int r = 0; r < 8; ++r) o[t][r] *= alpha[r];

    // C-layout P -> per-wave LDS slab -> A-layout fragment.
    bf16_t* pw = &sP[wave * 16 * LP];
#pragma unroll
    for (int r = 0; r < 8; ++r) {
      pw[(half * 8 + r) * LP + col16]      = (bf16_t)s0[r];
      pw[(half * 8 + r) * LP + 16 + col16] = (bf16_t)s1[r];
    }
    __syncthreads();   // visibility across lanes + compiler ordering

    const v16bf pa = load_frag_a(pw, LP, lane);
#pragma unroll
    for (int t = 0; t < 32; ++t) {
      // V^T rows (d, key): N-major B layout, contiguous keys per lane.
      const v16bf bv = load_frag_b(Vbh + (size_t)(t * 16) * S_ + k0, S_, lane);
      o[t] = wmma_bf16(pa, bv, o[t]);
    }
  }

  // Normalize and store into attn (B, S, H*D) layout for the final GEMM.
  bf16_t* outp = attn + ((size_t)(b * S_ + q0 + wave * 16)) * HD_ + (size_t)h * D_;
#pragma unroll
  for (int t = 0; t < 32; ++t) {
    const int col = t * 16 + col16;
#pragma unroll
    for (int r = 0; r < 8; ++r) {
      const int row = half * 8 + r;
      outp[(size_t)row * HD_ + col] = (bf16_t)(o[t][r] / lrow[r]);
    }
  }
}

// ---------------------------------------------------------------------------
// Kernel 3: out = attn(4096x4096, bf16) @ W_o(4096x512, f32), f32 output.
// ---------------------------------------------------------------------------
__global__ __launch_bounds__(128)
void out_proj_kernel(const bf16_t* __restrict__ Ag,
                     const float* __restrict__ Wo,
                     float* __restrict__ out) {
  const int tid  = threadIdx.x;
  const int lane = tid & 31;
  const int wave = tid >> 5;
  const int m0   = blockIdx.x * 64;     // B*S rows
  const int n0   = blockIdx.y * 64;     // D cols

  constexpr int LT = 40;
  __shared__ bf16_t sA[64 * LT];
  __shared__ bf16_t sB[64 * LT];

  v8f acc[4];
#pragma unroll
  for (int t = 0; t < 4; ++t) acc[t] = vzero();

  for (int k0 = 0; k0 < HD_; k0 += 32) {
    // A tile 64x32 bf16: 128 x 8-element (16B) chunks at cols 0/16, one per thread.
    {
      const int r = tid >> 1;
      const int c = (tid & 1) * 16;
      *(uint4*)&sA[r * LT + c] =
          *(const uint4*)&Ag[(size_t)(m0 + r) * HD_ + k0 + c];
    }
    // W_o tile 32x64 f32, coalesced float4 reads, N-major stores.
#pragma unroll
    for (int it = 0; it < 4; ++it) {
      const int i  = tid + it * 128;
      const int kk = i >> 4;
      const int n4 = (i & 15) * 4;
      const float4 v = *(const float4*)&Wo[(size_t)(k0 + kk) * D_ + n0 + n4];
      sB[(n4 + 0) * LT + kk] = (bf16_t)v.x;
      sB[(n4 + 1) * LT + kk] = (bf16_t)v.y;
      sB[(n4 + 2) * LT + kk] = (bf16_t)v.z;
      sB[(n4 + 3) * LT + kk] = (bf16_t)v.w;
    }
    __syncthreads();
    const v16bf af = load_frag_a(&sA[wave * 16 * LT], LT, lane);
#pragma unroll
    for (int t = 0; t < 4; ++t) {
      const v16bf bfg = load_frag_b(&sB[t * 16 * LT], LT, lane);
      acc[t] = wmma_bf16(af, bfg, acc[t]);
    }
    __syncthreads();
  }

  const int half  = lane >> 4;
  const int col16 = lane & 15;
#pragma unroll
  for (int t = 0; t < 4; ++t) {
    const int col = n0 + t * 16 + col16;
#pragma unroll
    for (int r = 0; r < 8; ++r) {
      const int row = m0 + wave * 16 + half * 8 + r;
      out[(size_t)row * D_ + col] = acc[t][r];
    }
  }
}

// ---------------------------------------------------------------------------
extern "C" void kernel_launch(void* const* d_in, const int* in_sizes, int n_in,
                              void* d_out, int out_size, void* d_ws, size_t ws_size,
                              hipStream_t stream) {
  const float* q  = (const float*)d_in[0];
  const float* Wq = (const float*)d_in[1];
  const float* Wk = (const float*)d_in[2];
  const float* Wv = (const float*)d_in[3];
  const float* Wo = (const float*)d_in[4];
  float* out = (float*)d_out;

  const size_t per = (size_t)B_ * H_ * S_ * D_;  // 16,777,216 elems
  bf16_t* qh   = (bf16_t*)d_ws;                  // needs 4*per*2 = ~134 MB
  bf16_t* kh   = qh + per;
  bf16_t* vht  = kh + per;                        // (bh, d, s) layout
  bf16_t* attn = vht + per;

  dim3 g1(S_ / 64, D_ / 64, 3 * B_ * H_);        // 32 x 8 x 48
  qkv_rope_kernel<<<g1, 128, 0, stream>>>(q, Wq, Wk, Wv, qh, kh, vht);

  dim3 g2(S_ / 64, B_ * H_);                     // 32 x 16
  flash_attn_kernel<<<g2, 128, 0, stream>>>(qh, kh, vht, attn);

  dim3 g3((B_ * S_) / 64, D_ / 64);              // 64 x 8
  out_proj_kernel<<<g3, 128, 0, stream>>>(attn, Wo, out);
}
